// TemporalAttention_68092411511607
// MI455X (gfx1250) — matrix operand order repaired and checked by reading, not compiled, not run on hardware
//
#include <hip/hip_runtime.h>
#include <stdint.h>

// ---------------------------------------------------------------------------
// CDNA5 (gfx1250) fused temporal attention, bf16 WMMA 16x16x32, fp32 accum.
// Kernel B LDS fill uses the Tensor Data Mover (tensor_load_to_lds).
// ---------------------------------------------------------------------------

typedef __attribute__((ext_vector_type(16))) __bf16 v16bf;
typedef __attribute__((ext_vector_type(8)))  float  v8f;
typedef __attribute__((ext_vector_type(4)))  uint32_t v4u;
typedef __attribute__((ext_vector_type(8)))  int      v8i;
typedef __attribute__((ext_vector_type(4)))  int      v4i;

union Frag {
    v16bf    v;
    uint4    q[2];
    uint32_t u[8];
    uint16_t h[16];
};

__device__ __forceinline__ uint16_t f2bf(float f) {
    union { float f; uint32_t u; } cv; cv.f = f;
    uint32_t r = cv.u + 0x7FFFu + ((cv.u >> 16) & 1u);   // round-to-nearest-even
    return (uint16_t)(r >> 16);
}
__device__ __forceinline__ uint32_t pack2(float lo, float hi) {
    return (uint32_t)f2bf(lo) | ((uint32_t)f2bf(hi) << 16);
}
__device__ __forceinline__ v8f vzero() {
    v8f z; for (int i = 0; i < 8; ++i) z[i] = 0.0f; return z;
}
// wave-private LDS staging ordering (stores before loads within a wave)
__device__ __forceinline__ void wave_lds_sync() {
    asm volatile("s_wait_dscnt 0x0" ::: "memory");
}

// A-fragment (16x32 bf16): lane<16 -> K = kbase+0..15
//                          lane>=16 -> K = kbase+16..31
__device__ __forceinline__ Frag ldsAfrag(const uint16_t* base, int stride,
                                         int m, int kbase, int half) {
    Frag fr;
    const uint16_t* p = base + m * stride + kbase + half * 16;
    fr.q[0] = *(const uint4*)(p);
    fr.q[1] = *(const uint4*)(p + 8);
    return fr;
}
// B-fragment (32x16 bf16): lane<16 -> K = kbase+0..15 ; lane>=16 -> kbase+16..31
// source rows are columns of the weight (pre-transposed, row stride 512)
__device__ __forceinline__ Frag glbBfrag(const uint16_t* base, int n,
                                         int kbase, int half) {
    Frag fr;
    const uint16_t* p = base + (size_t)n * 512 + kbase + half * 16;
    fr.q[0] = *(const uint4*)(p);
    fr.q[1] = *(const uint4*)(p + 8);
    return fr;
}
__device__ __forceinline__ Frag ldsBfrag(const uint16_t* base, int stride,
                                         int n, int kbase, int half) {
    Frag fr;
    const uint16_t* p = base + n * stride + kbase + half * 16;
    fr.q[0] = *(const uint4*)(p);
    fr.q[1] = *(const uint4*)(p + 8);
    return fr;
}
__device__ __forceinline__ v8f wmma_bf16(const Frag& a, const Frag& b, v8f c) {
    return __builtin_amdgcn_wmma_f32_16x16x32_bf16(
        /*neg_a*/false, a.v, /*neg_b*/false, b.v,
        /*c_mod*/(short)0, c, /*reuse_a*/false, /*reuse_b*/false);
}

// ---------------------------------------------------------------------------
// Prep: transpose + f32->bf16 convert:  dst[c][r] = bf16(src[r][c])
// ---------------------------------------------------------------------------
__global__ void transpose_bf16(const float* __restrict__ src,
                               uint16_t* __restrict__ dst, int R, int C) {
    int idx = blockIdx.x * blockDim.x + threadIdx.x;
    if (idx >= R * C) return;
    int c = idx / R, r = idx - c * R;
    dst[(size_t)c * R + r] = f2bf(src[(size_t)r * C + c]);
}

// ---------------------------------------------------------------------------
// Kernel A: gather-transpose x -> LDS, qkv GEMM, attention (softmax+bias),
// write per-head attention output (bf16) to workspace.
// Grid: 512 WGs (b=2, h=32, 8 blocks of 4 w). Block: 128 threads = 4 waves.
// ---------------------------------------------------------------------------
__global__ __launch_bounds__(128) void ta_qkv_attn(
    const float* __restrict__ x, const float* __restrict__ pos_bias,
    const uint16_t* __restrict__ wqkvT, uint16_t* __restrict__ attn_out)
{
    __shared__ __align__(16) uint16_t xs[64][512];       // 4 pixels x 16 frames x 512c
    __shared__ __align__(16) uint16_t qs[4][4][16][64];  // per-wave q staging
    __shared__ __align__(16) uint16_t ks[4][4][16][64];  // per-wave k staging
    __shared__ __align__(16) uint16_t as_[4][4][16][32]; // attn, K 16..31 stays zero
    __shared__ float pb[2048];                           // pos_bias (8x16x16)

    const int tid   = threadIdx.x;
    const int wv    = tid >> 5;
    const int lane  = tid & 31;
    const int mlane = lane & 15;
    const int half  = lane >> 4;

    const int wg    = blockIdx.x;
    const int wblk  = wg & 7;
    const int h     = (wg >> 3) & 31;
    const int b     = wg >> 8;
    const int gpix0 = (b * 32 + h) * 32 + wblk * 4;

    // ---- load x slab: xs[lp*16+f][c] = x[b][c][f][h][w0+lp] ----
    {
        const int lp = tid & 3, e0 = tid >> 2;           // 4 lanes share (c,f)
        for (int i = 0; i < 256; ++i) {
            int pair = e0 + (i << 5);                    // 0..8191
            int f = pair & 15, c = pair >> 4;
            float v = x[((((size_t)b * 512 + c) * 16 + f) * 32 + h) * 32
                        + (wblk * 4 + lp)];
            xs[lp * 16 + f][c] = f2bf(v);
        }
    }
    for (int i = tid; i < 2048; i += 128) pb[i] = pos_bias[i];
    { uint32_t* az = (uint32_t*)as_;
      for (int i = tid; i < 4 * 4 * 16 * 32 / 2; i += 128) az[i] = 0u; }
    __syncthreads();

    for (int hh = 0; hh < 2; ++hh) {
        const int hd = wv + 4 * hh;                      // wave's head this pass

        // ---- Q (mat 0, scaled) and K (mat 1) projections ----
        for (int mat = 0; mat < 2; ++mat) {
            uint16_t (*stg)[16][64] = mat ? ks[wv] : qs[wv];
            const int ncolbase = mat * 512 + hd * 64;
            const float sc = mat ? 1.0f : 0.125f;        // SCALE = 64^-0.5
            for (int nt = 0; nt < 4; ++nt) {
                v8f acc[4];
                for (int p = 0; p < 4; ++p) acc[p] = vzero();
                for (int kk = 0; kk < 16; ++kk) {
                    Frag bf = glbBfrag(wqkvT, ncolbase + nt * 16 + mlane,
                                       kk * 32, half);
                    for (int p = 0; p < 4; ++p) {
                        Frag af = ldsAfrag(&xs[p * 16][0], 512, mlane,
                                           kk * 32, half);
                        acc[p] = wmma_bf16(af, bf, acc[p]);
                    }
                }
                for (int p = 0; p < 4; ++p)
                    for (int j = 0; j < 8; ++j)
                        stg[p][j + half * 8][nt * 16 + mlane] =
                            f2bf(acc[p][j] * sc);
            }
        }
        wave_lds_sync();

        // ---- sim = q k^T (+bias), softmax over j, stage attn as bf16 ----
        for (int p = 0; p < 4; ++p) {
            v8f s = vzero();
            for (int kk2 = 0; kk2 < 2; ++kk2) {
                Frag af = ldsAfrag(&qs[wv][p][0][0], 64, mlane, kk2 * 32, half);
                Frag bf = ldsBfrag(&ks[wv][p][0][0], 64, mlane, kk2 * 32, half);
                s = wmma_bf16(af, bf, s);
            }
            for (int j = 0; j < 8; ++j) {
                int i = j + half * 8;                    // sim row (query frame)
                float v = s[j] + pb[(hd * 16 + i) * 16 + mlane];
                float m = v;                             // row-max across 16 lanes
                for (int o = 1; o < 16; o <<= 1) m = fmaxf(m, __shfl_xor(m, o, 32));
                float e = __expf(v - m);
                float sum = e;
                for (int o = 1; o < 16; o <<= 1) sum += __shfl_xor(sum, o, 32);
                as_[wv][p][i][mlane] = f2bf(e / sum);
            }
        }
        wave_lds_sync();

        // ---- V projection, then out = attn @ v, store bf16 to workspace ----
        const int vcolbase = 1024 + hd * 64;
        for (int nt = 0; nt < 4; ++nt) {
            v8f vacc[4];
            for (int p = 0; p < 4; ++p) vacc[p] = vzero();
            for (int kk = 0; kk < 16; ++kk) {
                Frag bf = glbBfrag(wqkvT, vcolbase + nt * 16 + mlane,
                                   kk * 32, half);
                for (int p = 0; p < 4; ++p) {
                    Frag af = ldsAfrag(&xs[p * 16][0], 512, mlane, kk * 32, half);
                    vacc[p] = wmma_bf16(af, bf, vacc[p]);
                }
            }
            for (int p = 0; p < 4; ++p) {
                // v accum (C-layout: lane=d, VGPR=frame) -> B-fragment in regs:
                // lane<16 supplies K=frames 0..15 (upper 8 via cross-half shfl),
                // lane>=16 (K 16..31) is zero, matching zero-padded attn K.
                float up[8];
                for (int j = 0; j < 8; ++j)
                    up[j] = __shfl_xor(vacc[p][j], 16, 32);
                Frag vb;
                for (int r2 = 0; r2 < 4; ++r2) {
                    uint32_t lo4 = pack2(vacc[p][2 * r2], vacc[p][2 * r2 + 1]);
                    uint32_t hi4 = pack2(up[2 * r2],      up[2 * r2 + 1]);
                    vb.u[r2]     = half ? 0u : lo4;
                    vb.u[r2 + 4] = half ? 0u : hi4;
                }
                Frag af = ldsAfrag(&as_[wv][p][0][0], 32, mlane, 0, half);
                v8f o = wmma_bf16(af, vb, vzero());
                uint16_t* orow = attn_out + (size_t)(gpix0 + p) * 16 * 512
                                 + hd * 64 + nt * 16 + mlane;
                for (int j = 0; j < 8; ++j)
                    orow[(size_t)(j + half * 8) * 512] = f2bf(o[j]);
            }
        }
    }
}

// ---------------------------------------------------------------------------
// Kernel B: out = attn_out (bf16) @ w_out, scatter to (b,c,f,h,w) f32.
// LDS slab filled by the Tensor Data Mover (single 64KB contiguous DMA).
// Grid: 512 WGs; block 128 threads = 4 waves; wave wv owns pixel wv.
// ---------------------------------------------------------------------------
__global__ __launch_bounds__(128) void ta_proj(
    const uint16_t* __restrict__ attn_out, const uint16_t* __restrict__ woutT,
    float* __restrict__ out)
{
    __shared__ __align__(16) uint16_t aslab[64][512];    // 64 KB
    __shared__ __align__(16) float obuf[16][16][4];      // [cout][f][w] 4 KB

    const int tid   = threadIdx.x;
    const int wv    = tid >> 5;
    const int lane  = tid & 31;
    const int mlane = lane & 15;
    const int half  = lane >> 4;

    const int wg    = blockIdx.x;
    const int wblk  = wg & 7;
    const int h     = (wg >> 3) & 31;
    const int b     = wg >> 8;
    const int gpix0 = (b * 32 + h) * 32 + wblk * 4;

    // ---- TDM: DMA 64 contiguous rows (64KB) of attn_out into LDS ----------
    // D# per CDNA5 ISA: 1-D tile of 8192 8-byte elements.
    if (wv == 0) {                                  // whole wave 0: scalar branch
        uint64_t ga = (uint64_t)(uintptr_t)(attn_out + (size_t)gpix0 * 16 * 512);
        uint32_t lds = (uint32_t)(uintptr_t)&aslab[0][0];   // LDS byte offset

        v4u g0;                                      // group0 (128b)
        g0[0] = 1u;                                  // count=1, user mode
        g0[1] = lds;                                 // lds_addr
        g0[2] = (uint32_t)ga;                        // global_addr[31:0]
        g0[3] = (uint32_t)((ga >> 32) & 0x1FFFFFFu)  // global_addr[56:32]
                | (2u << 30);                        // type = 2 ("image")

        v8i g1;                                      // group1 (256b)
        g1[0] = (int)(3u << 16);                     // mask=0, data_size=8B
        g1[1] = (int)(0x2000u << 16);                // tensor_dim0 = 8192 (lo16)
        g1[2] = (int)(1u << 16);                     // dim0 hi=0, tensor_dim1 = 1
        g1[3] = (int)(0x2000u << 16);                // tile_dim0 = 8192
        g1[4] = (int)((1u) | (1u << 16));            // tile_dim1 = 1, tile_dim2 = 1
        g1[5] = (int)0x2000u;                        // tensor_dim0_stride = 8192
        g1[6] = (int)(0x2000u << 16);                // stride0 hi=0, stride1 lo=8192
        g1[7] = 0;                                   // stride1 hi = 0

        v4i g2;                                      // group2: tensor_dim2=1
        g2[0] = 1; g2[1] = 0; g2[2] = 0; g2[3] = 0;
        v4i g3;                                      // group3: zeros
        g3[0] = 0; g3[1] = 0; g3[2] = 0; g3[3] = 0;
        v8i g4;                                      // trailing group (unused)
        for (int i = 0; i < 8; ++i) g4[i] = 0;

        __builtin_amdgcn_tensor_load_to_lds(g0, g1, g2, g3, g4, 0);
        __builtin_amdgcn_s_wait_tensorcnt(0);
    }
    __syncthreads();

    for (int nt = 0; nt < 32; ++nt) {
        if (nt + 1 < 32)   // hint next weight tile toward the caches
            __builtin_prefetch(woutT + (size_t)((nt + 1) * 16 + mlane) * 512, 0, 1);
        v8f acc = vzero();
        for (int kk = 0; kk < 16; ++kk) {
            Frag af = ldsAfrag(&aslab[wv * 16][0], 512, mlane, kk * 32, half);
            Frag bf = glbBfrag(woutT, nt * 16 + mlane, kk * 32, half);
            acc = wmma_bf16(af, bf, acc);
        }
        for (int j = 0; j < 8; ++j)
            obuf[mlane][j + half * 8][wv] = acc[j];
        __syncthreads();
        // cooperative w-contiguous store: 256 (cout,f) pairs, 16B each
        for (int i = 0; i < 2; ++i) {
            int pid = tid + i * 128;
            int cl = pid >> 4, f = pid & 15;
            float4 v4 = *(const float4*)&obuf[cl][f][0];
            int cout = nt * 16 + cl;
            float* dst = out + ((((size_t)b * 512 + cout) * 16 + f) * 32 + h) * 32
                         + wblk * 4;
            *(float4*)dst = v4;
        }
        __syncthreads();
    }
}

// ---------------------------------------------------------------------------
extern "C" void kernel_launch(void* const* d_in, const int* in_sizes, int n_in,
                              void* d_out, int out_size, void* d_ws, size_t ws_size,
                              hipStream_t stream) {
    const float* x        = (const float*)d_in[0];
    const float* pos_bias = (const float*)d_in[1];
    const float* w_qkv    = (const float*)d_in[2];   // (512, 1536)
    const float* w_out    = (const float*)d_in[3];   // (512, 512)
    float*       out      = (float*)d_out;

    uint16_t* wqkvT = (uint16_t*)d_ws;               // [1536][512] bf16
    uint16_t* woutT = wqkvT + 1536 * 512;            // [512][512]  bf16
    uint16_t* attn  = woutT + 512 * 512;             // [32768][512] bf16 (32 MB)

    transpose_bf16<<<(512 * 1536 + 255) / 256, 256, 0, stream>>>(w_qkv, wqkvT, 512, 1536);
    transpose_bf16<<<(512 * 512 + 255) / 256, 256, 0, stream>>>(w_out, woutT, 512, 512);
    ta_qkv_attn<<<512, 128, 0, stream>>>(x, pos_bias, wqkvT, attn);
    ta_proj<<<512, 128, 0, stream>>>(attn, woutT, out);
}
